// Attention_17471926960639
// MI455X (gfx1250) — matrix operand order repaired and checked by reading, not compiled, not run on hardware
//
#include <hip/hip_runtime.h>

typedef __attribute__((ext_vector_type(16))) _Float16 v16h;
typedef __attribute__((ext_vector_type(8)))  float    v8f;

#define BB 128
#define TT 2048
#define HH 512
#define NTILE 32   // HH/16  (n tiles of 16 output cols g)
#define KSTEP 16   // HH/32  (k steps of 32)
#define NPAIR (NTILE / 2)          // 16 n-tile pairs
#define PAIR_HALFS (2 * KSTEP * 32 * 16)   // 16384 f16 = 32 KB per pair

// ---------------------------------------------------------------------------
// Kernel 0: repack w2 (f32, [g][h] row-major) into f16 WMMA B-fragment layout.
// Fragment f = nt*KSTEP + ks holds B(K=32 x N=16) for k-block ks, n-block nt.
// Lane L: N = L%16, element e: K = (e&7) + 8*(L/16) + 16*(e/8).  B[k][n] = w2[g=nt*16+n][h=ks*32+k]
// Linear layout: w2p[(f*32 + lane)*16 + e]  -> each lane's 16 halves contiguous.
// ---------------------------------------------------------------------------
__global__ __launch_bounds__(256) void pack_w2_kernel(const float* __restrict__ w2,
                                                      _Float16* __restrict__ w2p) {
    int id   = blockIdx.x * 256 + threadIdx.x;      // 0 .. 512*512-1
    int e    = id & 15;
    int lane = (id >> 4) & 31;
    int f    = id >> 9;                             // 0..511
    int ks   = f & (KSTEP - 1);
    int nt   = f / KSTEP;
    int g    = nt * 16 + (lane & 15);
    int h    = ks * 32 + (e & 7) + 8 * (lane >> 4) + 16 * (e >> 3);
    w2p[id]  = (_Float16)w2[g * HH + h];
}

// ---------------------------------------------------------------------------
// Kernel 1: q_h = relu(query @ w1^T)   (B x H, tiny)
// ---------------------------------------------------------------------------
__global__ __launch_bounds__(256) void qh_kernel(const float* __restrict__ q,
                                                 const float* __restrict__ w1,
                                                 float* __restrict__ qh) {
    int idx = blockIdx.x * 256 + threadIdx.x;       // b*HH + g
    int b = idx >> 9, g = idx & (HH - 1);
    const float4* qp = (const float4*)(q + b * HH);
    const float4* wp = (const float4*)(w1 + g * HH);
    float acc = 0.f;
    #pragma unroll 4
    for (int i = 0; i < HH / 4; ++i) {
        float4 a = qp[i], w = wp[i];
        acc += a.x * w.x + a.y * w.y + a.z * w.z + a.w * w.w;
    }
    qh[idx] = fmaxf(acc, 0.f);
}

__device__ __forceinline__ float fast_tanh(float x) {
    // tanh(|x|) = 1 - 2/(exp(2|x|)+1), exp via v_exp_f32
    float ax = __builtin_fabsf(x);
    float e  = __builtin_amdgcn_exp2f(ax * 2.8853900817779268f); // 2*log2(e)
    float t  = 1.f - 2.f / (e + 1.f);
    return x < 0.f ? -t : t;
}

// ---------------------------------------------------------------------------
// Kernel 2 (hot): per 16-row t-strip, compute scores[b][t] =
//   sum_g tanh(qh[b][g] + relu(sum_h key[b][t][h]*w2[g][h])) * w_out[g]
// 8 waves / block, one strip per wave. A (16x512 of key, f16) lives in 128
// VGPRs. B n-tile pairs (32 KB) are staged into LDS with async copies
// (double buffered, 64 KB), shared by all 8 waves -> 8x less L2 traffic.
// Four independent WMMA chains (2 n-tiles x 2 K-halves) for deep pipelining.
// ---------------------------------------------------------------------------
__global__ __launch_bounds__(256) void score_kernel(const float* __restrict__ key,
                                                    const _Float16* __restrict__ w2p,
                                                    const float* __restrict__ qh,
                                                    const float* __restrict__ w_out,
                                                    float* __restrict__ scores) {
    __shared__ __align__(128) _Float16 bbuf[2][PAIR_HALFS];  // 2 x 32 KB

    int tid   = threadIdx.x;
    int lane  = tid & 31;
    int wave  = tid >> 5;
    int strip = blockIdx.x * 8 + wave;              // 0 .. BB*TT/16 - 1
    int b     = strip >> 7;                         // TT/16 = 128 strips / batch
    int t0    = (strip & 127) * 16;
    int m     = lane & 15;
    int half  = lane >> 4;

    // ---- load A fragments: row M=m of key strip, ISA 16-bit A layout ----
    // element e -> K = e + 8*((e>=8) + half)
    const float* arow = key + ((size_t)b * TT + t0 + m) * HH;
    v16h a[KSTEP];
    #pragma unroll
    for (int ks = 0; ks < KSTEP; ++ks) {
        const float* p = arow + ks * 32;
        int o1 = 8 * half;          // e = 0..7
        int o2 = 16 + 8 * half;     // e = 8..15
        float4 r0 = *(const float4*)(p + o1);
        float4 r1 = *(const float4*)(p + o1 + 4);
        float4 r2 = *(const float4*)(p + o2);
        float4 r3 = *(const float4*)(p + o2 + 4);
        v16h av;
        av[0]  = (_Float16)r0.x; av[1]  = (_Float16)r0.y;
        av[2]  = (_Float16)r0.z; av[3]  = (_Float16)r0.w;
        av[4]  = (_Float16)r1.x; av[5]  = (_Float16)r1.y;
        av[6]  = (_Float16)r1.z; av[7]  = (_Float16)r1.w;
        av[8]  = (_Float16)r2.x; av[9]  = (_Float16)r2.y;
        av[10] = (_Float16)r2.z; av[11] = (_Float16)r2.w;
        av[12] = (_Float16)r3.x; av[13] = (_Float16)r3.y;
        av[14] = (_Float16)r3.z; av[15] = (_Float16)r3.w;
        a[ks] = av;
    }

    // ---- async-stage helper: pair p -> bbuf[buf] (32 KB, 128 B per thread) ----
    // global_load_async_to_lds_b128: VDST = LDS byte offset (low 32 bits of the
    // generic shared-pointer ARE the LDS offset per the aperture rules),
    // VADDR = 64-bit global address. Tracked by ASYNCcnt.
    auto stage_pair = [&](int p, int buf) {
        const char* g = (const char*)(w2p + (size_t)p * PAIR_HALFS) + tid * 128;
        unsigned    l = (unsigned)(size_t)((const char*)&bbuf[buf][0] + tid * 128);
        #pragma unroll
        for (int j = 0; j < 8; ++j) {
            asm volatile("global_load_async_to_lds_b128 %0, %1, off"
                         :: "v"(l + j * 16), "v"(g + j * 16)
                         : "memory");
        }
    };

    float sc[8];
    #pragma unroll
    for (int r = 0; r < 8; ++r) sc[r] = 0.f;

    // prologue: stage pair 0
    stage_pair(0, 0);
    asm volatile("s_wait_asynccnt 0x0" ::: "memory");
    __syncthreads();

    #pragma unroll 1
    for (int p = 0; p < NPAIR; ++p) {
        // prefetch next pair into the other buffer (issue only; that buffer was
        // drained by everyone at the end-of-previous-iteration barrier)
        if (p + 1 < NPAIR) stage_pair(p + 1, (p + 1) & 1);

        // ---- compute: FOUR independent WMMA chains:
        //      (n-tile 2p, K-low) (n-tile 2p+1, K-low)
        //      (n-tile 2p, K-hi ) (n-tile 2p+1, K-hi )
        //      K-halves summed in f32 afterwards (exact same reduction).
        const v16h* lb = (const v16h*)&bbuf[p & 1][0] + lane;
        v8f c0 = {}, c1 = {}, d0 = {}, d1 = {};
        #pragma unroll
        for (int ks = 0; ks < KSTEP / 2; ++ks) {
            v16h b0 = lb[ks * 32];
            v16h b1 = lb[ks * 32 + KSTEP * 32];
            v16h b2 = lb[(ks + KSTEP / 2) * 32];
            v16h b3 = lb[(ks + KSTEP / 2) * 32 + KSTEP * 32];
            c0 = __builtin_amdgcn_wmma_f32_16x16x32_f16(false, a[ks], false, b0,
                                                        (short)0, c0, false, false);
            c1 = __builtin_amdgcn_wmma_f32_16x16x32_f16(false, a[ks], false, b1,
                                                        (short)0, c1, false, false);
            d0 = __builtin_amdgcn_wmma_f32_16x16x32_f16(false, a[ks + KSTEP / 2], false, b2,
                                                        (short)0, d0, false, false);
            d1 = __builtin_amdgcn_wmma_f32_16x16x32_f16(false, a[ks + KSTEP / 2], false, b3,
                                                        (short)0, d1, false, false);
        }

        // epilogue: merge K-halves, relu + qh + tanh, dot with w_out
        int g0 = (2 * p) * 16 + m;
        int g1 = g0 + 16;
        float qv0 = qh[b * HH + g0], qv1 = qh[b * HH + g1];
        float wv0 = w_out[g0],       wv1 = w_out[g1];
        #pragma unroll
        for (int r = 0; r < 8; ++r) {
            float x0 = fmaxf(c0[r] + d0[r], 0.f) + qv0;
            float x1 = fmaxf(c1[r] + d1[r], 0.f) + qv1;
            sc[r] += fast_tanh(x0) * wv0 + fast_tanh(x1) * wv1;
        }

        // single barrier: guarantees next buffer is filled AND current buffer
        // is drained by all waves before it gets overwritten next iteration
        asm volatile("s_wait_asynccnt 0x0" ::: "memory");
        __syncthreads();
    }

    // ---- reduce over the 16 lanes of each half-group (different g, same rows) ----
    #pragma unroll
    for (int r = 0; r < 8; ++r) {
        float v = sc[r];
        v += __shfl_xor(v, 1, 32);
        v += __shfl_xor(v, 2, 32);
        v += __shfl_xor(v, 4, 32);
        v += __shfl_xor(v, 8, 32);
        sc[r] = v;
    }
    // rows: VGPR r holds M = r + 8*half; one lane per half-group writes 8 rows
    if (m == 0) {
        float* srow = scores + (size_t)b * TT + t0 + 8 * half;
        #pragma unroll
        for (int r = 0; r < 8; ++r) srow[r] = sc[r];
    }
}

// ---------------------------------------------------------------------------
// Kernel 3: softmax over T + attn-weighted sum of key  (memory-bound pass)
// ---------------------------------------------------------------------------
__global__ __launch_bounds__(256) void out_kernel(const float* __restrict__ key,
                                                  const float* __restrict__ scores,
                                                  float* __restrict__ out) {
    __shared__ float sm[TT];
    __shared__ float red[256];
    int b = blockIdx.x, tid = threadIdx.x;
    const float* srow = scores + (size_t)b * TT;

    float mx = -1e30f;
    for (int t = tid; t < TT; t += 256) {
        float v = srow[t];
        sm[t] = v;
        mx = fmaxf(mx, v);
    }
    red[tid] = mx; __syncthreads();
    for (int s = 128; s > 0; s >>= 1) {
        if (tid < s) red[tid] = fmaxf(red[tid], red[tid + s]);
        __syncthreads();
    }
    mx = red[0]; __syncthreads();

    float sum = 0.f;
    for (int t = tid; t < TT; t += 256) {
        float e = __builtin_amdgcn_exp2f((sm[t] - mx) * 1.4426950408889634f);
        sm[t] = e;
        sum += e;
    }
    red[tid] = sum; __syncthreads();
    for (int s = 128; s > 0; s >>= 1) {
        if (tid < s) red[tid] += red[tid + s];
        __syncthreads();
    }
    float inv = 1.f / red[0];

    // each thread owns columns h = tid and tid+256; coalesced over key rows
    const float* kb = key + (size_t)b * TT * HH;
    float acc0 = 0.f, acc1 = 0.f;
    for (int t = 0; t < TT; ++t) {
        float w = sm[t];
        acc0 += w * kb[(size_t)t * HH + tid];
        acc1 += w * kb[(size_t)t * HH + tid + 256];
    }
    out[b * HH + tid]       = acc0 * inv;
    out[b * HH + tid + 256] = acc1 * inv;
}

// ---------------------------------------------------------------------------
extern "C" void kernel_launch(void* const* d_in, const int* in_sizes, int n_in,
                              void* d_out, int out_size, void* d_ws, size_t ws_size,
                              hipStream_t stream) {
    const float* query = (const float*)d_in[0];   // (B,H)
    const float* key   = (const float*)d_in[1];   // (B,T,H)
    const float* w1    = (const float*)d_in[2];   // (H,H)
    const float* w2    = (const float*)d_in[3];   // (H,H)
    const float* wout  = (const float*)d_in[4];   // (H,)
    float* out = (float*)d_out;

    char* ws = (char*)d_ws;
    _Float16* w2p   = (_Float16*)ws;                       // 512 KB
    float*    qh    = (float*)(ws + (512 << 10));          // 256 KB
    float*    score = (float*)(ws + (768 << 10));          // 1 MB

    // 1) repack w2 -> f16 B-fragments
    pack_w2_kernel<<<(HH * HH) / 256, 256, 0, stream>>>(w2, w2p);
    // 2) q_h = relu(q @ w1^T)
    qh_kernel<<<(BB * HH) / 256, 256, 0, stream>>>(query, w1, qh);
    // 3) fused GEMM + tanh-score (WMMA + async LDS staging): 8 waves / block
    score_kernel<<<(BB * TT / 16) / 8, 256, 0, stream>>>(key, w2p, qh, wout, score);
    // 4) softmax + weighted sum
    out_kernel<<<BB, 256, 0, stream>>>(key, score, out);
}
